// Big_MPNN_17703855194119
// MI455X (gfx1250) — matrix operand order repaired and checked by reading, not compiled, not run on hardware
//
#include <hip/hip_runtime.h>

// ---------------------------------------------------------------------------
// CDNA5 (gfx1250) wave32 WMMA implementation of the 3-pass MPNN.
// All GEMMs use v_wmma_f32_16x16x32_bf16 (bf16 operands, f32 accumulate).
// ---------------------------------------------------------------------------

typedef __attribute__((ext_vector_type(8)))  __bf16 bf16x8;
typedef __attribute__((ext_vector_type(16))) __bf16 v16bf;
typedef __attribute__((ext_vector_type(8)))  float  v8f;

#define DIM    128        // message size D
#define NNODE  16384      // B*N = 128*128
#define NBOND  7
#define NLAY   8
#define NTYPE  6
#define H3     384        // 3*D

// ---- fragment helpers (layouts per CDNA5 ISA 7.12.2, wave32) ---------------
// A frag (16x32 bf16): lane row = l16; halves0-7 at K= ks*32 + 8*hi + 0..7,
//                      halves8-15 at K= ks*32 + 16 + 8*hi + 0..7
// B frag (32x16 bf16): lane col = l16; same K layout as A frag
__device__ __forceinline__ v16bf load_frag16(const __bf16* p, int off0, int off1) {
  bf16x8 lo = *(const bf16x8*)(p + off0);
  bf16x8 hi = *(const bf16x8*)(p + off1);
  return __builtin_shufflevector(lo, hi, 0,1,2,3,4,5,6,7,8,9,10,11,12,13,14,15);
}

__device__ __forceinline__ v16bf load_fragA(const __bf16* row, int ks, int hi) {
  return load_frag16(row, ks * 32 + 8 * hi, ks * 32 + 8 * hi + 16);
}
__device__ __forceinline__ v16bf load_fragB(const __bf16* row, int ks, int hi) {
  return load_frag16(row, ks * 32 + 8 * hi, ks * 32 + 8 * hi + 16);
}

__device__ __forceinline__ v8f wmma_bf(v16bf a, v16bf b, v8f c) {
  return __builtin_amdgcn_wmma_f32_16x16x32_bf16(false, a, false, b, (short)0, c,
                                                 false, false);
}

__device__ __forceinline__ void store_v8f(float* p, v8f c) {
  float4 a = {c[0], c[1], c[2], c[3]};
  float4 b = {c[4], c[5], c[6], c[7]};
  *(float4*)(p)     = a;
  *(float4*)(p + 4) = b;
}

__device__ __forceinline__ bf16x8 cvt_bf8(v8f c) {
  bf16x8 o;
#pragma unroll
  for (int j = 0; j < 8; ++j) o[j] = (__bf16)c[j];
  return o;
}

// ---------------------------------------------------------------------------
// One-time: convert g / msg_W / gru weights to bf16 (total 16,777,216 elems)
// ---------------------------------------------------------------------------
__global__ void prep_w_kernel(const float* __restrict__ g,
                              const float* __restrict__ msgW,
                              const float* __restrict__ Wih,
                              const float* __restrict__ Whh,
                              __bf16* gbf, __bf16* msgWbf,
                              __bf16* Wihbf, __bf16* Whhbf) {
  const size_t NG = 14680064, NM = 917504, NI = 589824, NH = 589824;
  const size_t total = NG + NM + NI + NH;
  for (size_t i = (size_t)blockIdx.x * blockDim.x + threadIdx.x; i < total;
       i += (size_t)gridDim.x * blockDim.x) {
    if (i < NG)                gbf[i]              = (__bf16)g[i];
    else if (i < NG + NM)      msgWbf[i - NG]      = (__bf16)msgW[i - NG];
    else if (i < NG + NM + NI) Wihbf[i - NG - NM]  = (__bf16)Wih[i - NG - NM];
    else                       Whhbf[i - NG - NM - NI] = (__bf16)Whh[i - NG - NM - NI];
  }
}

// ---------------------------------------------------------------------------
// One-time: pad h (75 -> 128), bf16 copy, and per-node GRU type id
// ---------------------------------------------------------------------------
__global__ void prep_h_kernel(const float* __restrict__ h, float* hf,
                              __bf16* hbt, int* tid) {
  size_t i = (size_t)blockIdx.x * blockDim.x + threadIdx.x;  // exact NNODE*128
  int node = (int)(i >> 7), d = (int)(i & 127);
  float v = (d < 75) ? h[(size_t)node * 75 + d] : 0.f;
  hf[i]  = v;
  hbt[i] = (__bf16)v;
  if (d == 0) {
    int t = 5;
    if (v == 6.f) t = 0; else if (v == 7.f) t = 1; else if (v == 8.f) t = 2;
    else if (v == 9.f) t = 3; else if (v == 0.f) t = 4;
    tid[node] = t;
  }
}

// ---------------------------------------------------------------------------
// MLP layer:  Y[m][e] = act( X[m][:] . W[e][:] )  per bond type k.
// TRANSPOSE==0 (layers 0..6): A=W (LDS), B=X (regs), ReLU,
//                             out row-major xb[k][node][e]
// TRANSPOSE==1 (layer 7):     A=X (regs), B=W (LDS), no ReLU,
//                             out transposed xbT[k][b][d][m]
// Compile-time template => branch-free straight-line WMMA clauses.
// grid (128 node-blocks, 7 bonds), 256 thr = 8 waves, wave = 16-node tile
// ---------------------------------------------------------------------------
template <int TRANSPOSE>
__global__ void mlp_layer_kernel(const __bf16* __restrict__ Xin,
                                 const __bf16* __restrict__ Wall,
                                 __bf16* __restrict__ Xout,
                                 int layer, int inPerBond) {
  __shared__ __bf16 Wlds[DIM * DIM];  // 32 KB: W[k][layer], row-major [e][d]
  const int k  = blockIdx.y;
  const int bb = blockIdx.x;  // block of 128 nodes == molecule b
  {
    const uint4* src = (const uint4*)(Wall + (size_t)(k * NLAY + layer) * DIM * DIM);
    uint4* dst = (uint4*)Wlds;
    for (int i = threadIdx.x; i < DIM * DIM / 8; i += blockDim.x) dst[i] = src[i];
  }
  __syncthreads();

  const int wv = threadIdx.x >> 5, lane = threadIdx.x & 31;
  const int l16 = lane & 15, hi = lane >> 4;
  const size_t nodeRow = (size_t)(bb * 128 + wv * 16 + l16);
  const __bf16* xrow = Xin + (inPerBond ? (size_t)k * NNODE * DIM : 0) + nodeRow * DIM;

  v16bf xf[4];  // the wave's 16 X rows, full K=128, kept resident
#pragma unroll
  for (int ks = 0; ks < 4; ++ks)
    xf[ks] = TRANSPOSE ? load_fragA(xrow, ks, hi) : load_fragB(xrow, ks, hi);

#pragma unroll
  for (int et = 0; et < 8; ++et) {  // 8 output col tiles (e / d)
    v8f c = {0.f, 0.f, 0.f, 0.f, 0.f, 0.f, 0.f, 0.f};
    const __bf16* wrow = Wlds + (et * 16 + l16) * DIM;
#pragma unroll
    for (int ks = 0; ks < 4; ++ks) {
      if (TRANSPOSE) {
        c = wmma_bf(xf[ks], load_fragB(wrow, ks, hi), c);
      } else {
        c = wmma_bf(load_fragA(wrow, ks, hi), xf[ks], c);
      }
    }
    if (!TRANSPOSE) {
#pragma unroll
      for (int j = 0; j < 8; ++j) c[j] = fmaxf(c[j], 0.f);  // ReLU (layers 0..6)
      // Out[node][e]: lane col = node, 8 contiguous e values
      __bf16* op = Xout + (size_t)k * NNODE * DIM + nodeRow * DIM + et * 16 + 8 * hi;
      *(bf16x8*)op = cvt_bf8(c);
    } else {
      // xbT[k][b][d][m]: lane col = d, 8 contiguous m values
      __bf16* op = Xout + (((size_t)k * 128 + bb) * DIM + et * 16 + l16) * DIM
                        + wv * 16 + 8 * hi;
      *(bf16x8*)op = cvt_bf8(c);
    }
  }
}

// ---------------------------------------------------------------------------
// Aggregation: m_k[n][d] = sum_m g[b,k,n,m] * xb[k,b,m,d]
// A = xbT slice (LDS, shared across waves), B = g rows (global, L2-resident)
// accumulates bond types 0..5 -> m_non, type 6 -> m_uni.
// grid (128 b, 8 dtile), wave = ntile; fully unrolled bond loop.
// ---------------------------------------------------------------------------
__global__ void aggregate_kernel(const __bf16* __restrict__ gbf,
                                 const __bf16* __restrict__ xbT,
                                 float* mnon, float* muni,
                                 __bf16* mnonb, __bf16* munib) {
  __shared__ __bf16 xs[NBOND * 16 * DIM];  // 28 KB: xbT[k][dtile rows][m]
  const int b = blockIdx.x, dt = blockIdx.y;
  {
    uint4* dst = (uint4*)xs;
    for (int i = threadIdx.x; i < NBOND * 16 * DIM / 8; i += blockDim.x) {
      int k = i >> 8, rem = i & 255, r = rem >> 4, q = rem & 15;
      size_t row = ((size_t)k * 128 + b) * DIM + dt * 16 + r;
      dst[i] = ((const uint4*)xbT)[row * 16 + q];  // 16 uint4 per 128-elem row
    }
  }
  __syncthreads();

  const int wv = threadIdx.x >> 5, lane = threadIdx.x & 31;
  const int l16 = lane & 15, hi = lane >> 4;
  v8f accN = {0.f, 0.f, 0.f, 0.f, 0.f, 0.f, 0.f, 0.f};
  v8f accU = {0.f, 0.f, 0.f, 0.f, 0.f, 0.f, 0.f, 0.f};

#pragma unroll
  for (int k = 0; k < NBOND; ++k) {
    const __bf16* ar = xs + (k * 16 + l16) * DIM;  // A rows: d within tile
    const __bf16* br = gbf + (((size_t)b * NBOND + k) * 128 + wv * 16 + l16) * (size_t)DIM;
#pragma unroll
    for (int ks = 0; ks < 4; ++ks) {
      v16bf af  = load_fragA(ar, ks, hi);
      v16bf bf_ = load_fragB(br, ks, hi);
      if (k < 6) accN = wmma_bf(af, bf_, accN);   // compile-time select
      else       accU = wmma_bf(af, bf_, accU);
    }
  }
  size_t node = (size_t)b * 128 + wv * 16 + l16;
  int dbase = dt * 16 + 8 * hi;
  store_v8f(mnon + node * DIM + dbase, accN);
  store_v8f(muni + node * DIM + dbase, accU);
  *(bf16x8*)(mnonb + node * DIM + dbase) = cvt_bf8(accN);
  *(bf16x8*)(munib + node * DIM + dbase) = cvt_bf8(accU);
}

// ---------------------------------------------------------------------------
// GRU gate GEMMs: G_z[node][col] = In_z[node][:] . Wset_z[tid(node)][col][:]
// All 6 types computed per tile; per-lane select by tid (lane's N index=node).
// grid (128 node-blocks, z=4: gx0,gh0,gx1,gh1)
// ---------------------------------------------------------------------------
__global__ void gru_matmul_kernel(const __bf16* __restrict__ hbt,
                                  const __bf16* __restrict__ mnonb,
                                  const __bf16* __restrict__ munib,
                                  const __bf16* __restrict__ WihBf,
                                  const __bf16* __restrict__ WhhBf,
                                  const int* __restrict__ tid, float* G) {
  const int z = blockIdx.y;
  const __bf16* In = (z == 1) ? mnonb : (z == 3) ? munib : hbt;
  const __bf16* W  = ((z & 1) ? WhhBf : WihBf) + ((z >= 2) ? (size_t)NTYPE * H3 * DIM : 0);
  float* Gout = G + (size_t)z * NNODE * H3;

  const int wv = threadIdx.x >> 5, lane = threadIdx.x & 31;
  const int l16 = lane & 15, hi = lane >> 4;
  const size_t node = (size_t)blockIdx.x * 128 + wv * 16 + l16;

  v16bf bfr[4];  // B frags: the wave's 16 input rows, resident
#pragma unroll
  for (int ks = 0; ks < 4; ++ks) bfr[ks] = load_fragB(In + node * DIM, ks, hi);
  const int t_lane = tid[node];

  for (int ct = 0; ct < 24; ++ct) {  // 384/16 col tiles
    v8f acc[NTYPE];
#pragma unroll
    for (int t = 0; t < NTYPE; ++t)
      acc[t] = (v8f){0.f, 0.f, 0.f, 0.f, 0.f, 0.f, 0.f, 0.f};
#pragma unroll
    for (int t = 0; t < NTYPE; ++t) {
      const __bf16* wr = W + ((size_t)t * H3 + ct * 16 + l16) * DIM;
#pragma unroll
      for (int ks = 0; ks < 4; ++ks)
        acc[t] = wmma_bf(load_fragA(wr, ks, hi), bfr[ks], acc[t]);
    }
    v8f r = t_lane == 0 ? acc[0] : t_lane == 1 ? acc[1] : t_lane == 2 ? acc[2]
          : t_lane == 3 ? acc[3] : t_lane == 4 ? acc[4] : acc[5];
    store_v8f(Gout + node * H3 + ct * 16 + 8 * hi, r);
  }
}

// ---------------------------------------------------------------------------
// Pointwise GRU (both variants) + combine: h' = gru0(m_non) + gru1(m_uni)
// ---------------------------------------------------------------------------
__global__ void gru_pointwise_kernel(const float* __restrict__ G,
                                     const float* __restrict__ mnon,
                                     const float* __restrict__ muni,
                                     const float* __restrict__ bih,
                                     const float* __restrict__ bhh,
                                     const int* __restrict__ tid,
                                     float* hf, __bf16* hbt) {
  size_t i = (size_t)blockIdx.x * blockDim.x + threadIdx.x;  // exact NNODE*128
  int node = (int)(i >> 7), d = (int)(i & 127);
  int t = tid[node];
  const size_t GS = (size_t)NNODE * H3;
  const float* gbase = G + (size_t)node * H3;
  float out = 0.f;
#pragma unroll
  for (int v = 0; v < 2; ++v) {
    const float* gx = gbase + (size_t)(2 * v) * GS;       // z=0 / z=2
    const float* gh = gbase + (size_t)(2 * v + 1) * GS;   // z=1 / z=3
    const float* bi = bih + ((size_t)v * NTYPE + t) * H3;
    const float* bh = bhh + ((size_t)v * NTYPE + t) * H3;
    float hx = v ? muni[i] : mnon[i];
    float xr = gx[d] + bi[d], xz = gx[128 + d] + bi[128 + d], xn = gx[256 + d] + bi[256 + d];
    float hr = gh[d] + bh[d], hz = gh[128 + d] + bh[128 + d], hn = gh[256 + d] + bh[256 + d];
    float r  = 1.f / (1.f + __expf(-(xr + hr)));
    float zz = 1.f / (1.f + __expf(-(xz + hz)));
    float n  = tanhf(xn + r * hn);
    out += (1.f - zz) * n + zz * hx;
  }
  hf[i]  = out;
  hbt[i] = (__bf16)out;
}

// ---------------------------------------------------------------------------
extern "C" void kernel_launch(void* const* d_in, const int* in_sizes, int n_in,
                              void* d_out, int out_size, void* d_ws, size_t ws_size,
                              hipStream_t stream) {
  (void)in_sizes; (void)n_in; (void)out_size; (void)ws_size;
  const float* g    = (const float*)d_in[0];
  const float* h    = (const float*)d_in[1];
  const float* msgW = (const float*)d_in[2];
  const float* Wih  = (const float*)d_in[3];
  const float* Whh  = (const float*)d_in[4];
  const float* bih  = (const float*)d_in[5];
  const float* bhh  = (const float*)d_in[6];

  char* ws = (char*)d_ws;
  size_t off = 0;
  auto alloc = [&](size_t bytes) -> char* {
    char* p = ws + off;
    off += (bytes + 255) & ~(size_t)255;
    return p;
  };
  __bf16* gbf    = (__bf16*)alloc((size_t)14680064 * 2);
  __bf16* msgWbf = (__bf16*)alloc((size_t)917504 * 2);
  __bf16* Wihbf  = (__bf16*)alloc((size_t)589824 * 2);
  __bf16* Whhbf  = (__bf16*)alloc((size_t)589824 * 2);
  float*  hf     = (float*) alloc((size_t)NNODE * DIM * 4);
  __bf16* hbt    = (__bf16*)alloc((size_t)NNODE * DIM * 2);
  int*    tid    = (int*)   alloc((size_t)NNODE * 4);
  __bf16* xbA    = (__bf16*)alloc((size_t)NBOND * NNODE * DIM * 2);
  __bf16* xbB    = (__bf16*)alloc((size_t)NBOND * NNODE * DIM * 2);
  float*  mnon   = (float*) alloc((size_t)NNODE * DIM * 4);
  float*  muni   = (float*) alloc((size_t)NNODE * DIM * 4);
  __bf16* mnonb  = (__bf16*)alloc((size_t)NNODE * DIM * 2);
  __bf16* munib  = (__bf16*)alloc((size_t)NNODE * DIM * 2);
  float*  G      = (float*) alloc((size_t)4 * NNODE * H3 * 4);

  prep_w_kernel<<<8192, 256, 0, stream>>>(g, msgW, Wih, Whh, gbf, msgWbf, Wihbf, Whhbf);
  prep_h_kernel<<<8192, 256, 0, stream>>>(h, hf, hbt, tid);

  for (int p = 0; p < 3; ++p) {
    // layers 0..6: normal (ReLU), ping-pong xbA/xbB; layer 7: transposed out
    for (int l = 0; l < 7; ++l) {
      const __bf16* in = (l == 0) ? hbt : ((l & 1) ? xbA : xbB);
      __bf16* out = (l & 1) ? xbB : xbA;
      mlp_layer_kernel<0><<<dim3(128, NBOND), 256, 0, stream>>>(
          in, msgWbf, out, l, (l == 0) ? 0 : 1);
    }
    mlp_layer_kernel<1><<<dim3(128, NBOND), 256, 0, stream>>>(
        xbA, msgWbf, xbB, 7, 1);
    aggregate_kernel<<<dim3(128, 8), 256, 0, stream>>>(gbf, xbB, mnon, muni, mnonb, munib);
    gru_matmul_kernel<<<dim3(128, 4), 256, 0, stream>>>(hbt, mnonb, munib, Wihbf, Whhbf, tid, G);
    gru_pointwise_kernel<<<8192, 256, 0, stream>>>(G, mnon, muni, bih, bhh, tid, hf, hbt);
  }
  hipMemcpyAsync(d_out, hf, (size_t)NNODE * DIM * 4, hipMemcpyDeviceToDevice, stream);
}
